// ScaledDotProduct_71554155151920
// MI455X (gfx1250) — matrix operand order repaired and checked by reading, not compiled, not run on hardware
//
#include <hip/hip_runtime.h>

typedef __attribute__((ext_vector_type(16))) __bf16 v16bf;
typedef __attribute__((ext_vector_type(8)))  __bf16 bf16x8;
typedef __attribute__((ext_vector_type(8)))  float  v8f;

// ---------------------------------------------------------------------------
// WMMA fragment loaders (wave32, V_WMMA_F32_16X16X32_BF16)
//
// A matrix 16x32 bf16, row-major source [M][K] (ld elements per row):
//   lane l<16 : M=l,    holds K[0..16) in elems 0..15
//   lane l>=16: M=l-16, holds K[16..32)
__device__ __forceinline__ v16bf load_a_frag(const __bf16* __restrict__ base,
                                             int ld, int k, int lane) {
  const int half = lane >> 4;
  const int m    = lane & 15;
  const __bf16* p = base + (size_t)m * (size_t)ld + (size_t)(k + half * 16);
  v16bf f;
  ((bf16x8*)&f)[0] = *(const bf16x8*)(p);
  ((bf16x8*)&f)[1] = *(const bf16x8*)(p + 8);
  return f;
}

// B matrix 32x16 bf16 (KxN). Source stored row-major as [N][K] (i.e. B^T):
//   lane l: N = l&15, holds 16 contiguous K values starting at k + (l>=16)*16
__device__ __forceinline__ v16bf load_b_frag(const __bf16* __restrict__ baseT,
                                             int ld, int k, int lane) {
  const int n    = lane & 15;
  const int koff = (lane >> 4) * 16;
  const __bf16* p = baseT + (size_t)n * (size_t)ld + (size_t)(k + koff);
  v16bf f;
  ((bf16x8*)&f)[0] = *(const bf16x8*)(p);
  ((bf16x8*)&f)[1] = *(const bf16x8*)(p + 8);
  return f;
}

__device__ __forceinline__ v8f wmma_bf16(v16bf a, v16bf b, v8f c) {
  return __builtin_amdgcn_wmma_f32_16x16x32_bf16(false, a, false, b,
                                                 (short)0, c, false, false);
}

// ---------------------------------------------------------------------------
// Precision conversion kernels
__global__ __launch_bounds__(256) void cvt_bf16(const float* __restrict__ in,
                                                __bf16* __restrict__ out, int n) {
  int i = blockIdx.x * 256 + threadIdx.x;
  if (i < n) out[i] = (__bf16)in[i];
}

// W [1024][3072] row-major -> Wt [3072][1024] bf16 (B^T layout for GEMM1)
__global__ __launch_bounds__(256) void cvt_Wt(const float* __restrict__ W,
                                              __bf16* __restrict__ Wt) {
  int i = blockIdx.x * 256 + threadIdx.x;   // 0 .. 1024*3072-1, coalesced read
  int c = i / 3072;
  int d = i - c * 3072;
  Wt[(size_t)d * 1024 + c] = (__bf16)W[i];
}

// ---------------------------------------------------------------------------
// 64x64-per-wave GEMM core: 4 A-frags x 4 B-frags -> 16 WMMAs per 16 b128 loads.
// Workgroup = 8 waves arranged 2 (M) x 4 (N)  =>  128 x 256 block tile.
// ---------------------------------------------------------------------------

// GEMM1: qkv = x @ W + b.  M=8192 (b*t), N=3072, K=1024.
// Writes Q,K row-major bf16 [b][2048][1024]; V transposed bf16 [b][1024][2048].
__global__ __launch_bounds__(256)
void qkv_gemm(const __bf16* __restrict__ xb, const __bf16* __restrict__ Wt,
              const float* __restrict__ bias,
              __bf16* __restrict__ Qb, __bf16* __restrict__ Kb,
              __bf16* __restrict__ Vt) {
  const int lane = threadIdx.x & 31;
  const int wave = threadIdx.x >> 5;
  const int wm = wave & 1, wn = wave >> 1;
  const int mBase = blockIdx.y * 128 + wm * 64;
  const int nBase = blockIdx.x * 256 + wn * 64;

  const v8f zero = {0.f, 0.f, 0.f, 0.f, 0.f, 0.f, 0.f, 0.f};
  v8f acc[4][4];
#pragma unroll
  for (int i = 0; i < 4; ++i)
#pragma unroll
    for (int j = 0; j < 4; ++j) acc[i][j] = zero;

  const __bf16* A0 = xb + (size_t)mBase * 1024;
  for (int k = 0; k < 1024; k += 32) {
    v16bf a[4];
#pragma unroll
    for (int i = 0; i < 4; ++i)
      a[i] = load_a_frag(A0 + (size_t)(i * 16) * 1024, 1024, k, lane);
#pragma unroll
    for (int j = 0; j < 4; ++j) {
      v16bf bf = load_b_frag(Wt + (size_t)(nBase + j * 16) * 1024, 1024, k, lane);
#pragma unroll
      for (int i = 0; i < 4; ++i) acc[i][j] = wmma_bf16(a[i], bf, acc[i][j]);
    }
  }

  const int half = lane >> 4, nl = lane & 15;
#pragma unroll
  for (int i = 0; i < 4; ++i)
#pragma unroll
    for (int j = 0; j < 4; ++j) {
      const int col = nBase + j * 16 + nl;
      const float bval = bias[col];
#pragma unroll
      for (int r = 0; r < 8; ++r) {
        const int row = mBase + i * 16 + half * 8 + r;
        const float v = acc[i][j][r] + bval;
        const int bidx = row >> 11;
        const int t = row & 2047;
        if (col < 1024)
          Qb[((size_t)bidx * 2048 + t) * 1024 + col] = (__bf16)v;
        else if (col < 2048)
          Kb[((size_t)bidx * 2048 + t) * 1024 + (col - 1024)] = (__bf16)v;
        else
          Vt[((size_t)bidx * 1024 + (col - 2048)) * 2048 + t] = (__bf16)v;
      }
    }
}

// ---------------------------------------------------------------------------
// GEMM2: S = Q @ K^T  (per batch, 2048x2048, K-dim 1024). B^T = K row-major.
__global__ __launch_bounds__(256)
void scores_gemm(const __bf16* __restrict__ Qb, const __bf16* __restrict__ Kb,
                 float* __restrict__ S) {
  // block tile: rows [by*128, +127], cols [bx*256, +255]; skip if fully masked
  if ((int)(2 * blockIdx.x) > (int)blockIdx.y) return;
  const int z = blockIdx.z;
  const __bf16* Q  = Qb + (size_t)z * 2048 * 1024;
  const __bf16* Km = Kb + (size_t)z * 2048 * 1024;
  float* Sp = S + (size_t)z * 2048 * 2048;

  const int lane = threadIdx.x & 31;
  const int wave = threadIdx.x >> 5;
  const int wm = wave & 1, wn = wave >> 1;
  const int mBase = blockIdx.y * 128 + wm * 64;
  const int nBase = blockIdx.x * 256 + wn * 64;

  const v8f zero = {0.f, 0.f, 0.f, 0.f, 0.f, 0.f, 0.f, 0.f};
  v8f acc[4][4];
#pragma unroll
  for (int i = 0; i < 4; ++i)
#pragma unroll
    for (int j = 0; j < 4; ++j) acc[i][j] = zero;

  const __bf16* A0 = Q + (size_t)mBase * 1024;
  for (int k = 0; k < 1024; k += 32) {
    v16bf a[4];
#pragma unroll
    for (int i = 0; i < 4; ++i)
      a[i] = load_a_frag(A0 + (size_t)(i * 16) * 1024, 1024, k, lane);
#pragma unroll
    for (int j = 0; j < 4; ++j) {
      v16bf bf = load_b_frag(Km + (size_t)(nBase + j * 16) * 1024, 1024, k, lane);
#pragma unroll
      for (int i = 0; i < 4; ++i) acc[i][j] = wmma_bf16(a[i], bf, acc[i][j]);
    }
  }

  const int half = lane >> 4, nl = lane & 15;
#pragma unroll
  for (int i = 0; i < 4; ++i)
#pragma unroll
    for (int j = 0; j < 4; ++j) {
      const int col = nBase + j * 16 + nl;
#pragma unroll
      for (int r = 0; r < 8; ++r) {
        const int row = mBase + i * 16 + half * 8 + r;
        Sp[(size_t)row * 2048 + col] = acc[i][j][r];
      }
    }
}

// ---------------------------------------------------------------------------
// Causal softmax per row: P[t, :t+1] = softmax(S[t, :t+1] / 32), rest zero.
__global__ __launch_bounds__(256)
void softmax_causal(const float* __restrict__ S, __bf16* __restrict__ P) {
  const int row = blockIdx.x;       // 0..8191 (= b*2048 + t)
  const int t = row & 2047;
  const float* s = S + (size_t)row * 2048;
  __bf16* p = P + (size_t)row * 2048;
  const int n = t + 1;
  const float scale = 0.03125f;     // 1/sqrt(1024)
  const int tid = threadIdx.x;
  __shared__ float red[8];

  float m = -1e30f;
  for (int i = tid; i < n; i += 256) m = fmaxf(m, s[i]);
#pragma unroll
  for (int off = 16; off; off >>= 1) m = fmaxf(m, __shfl_xor(m, off, 32));
  if ((tid & 31) == 0) red[tid >> 5] = m;
  __syncthreads();
  m = red[0];
#pragma unroll
  for (int w = 1; w < 8; ++w) m = fmaxf(m, red[w]);
  m *= scale;                       // scale > 0 => max(scale*s) = scale*max(s)
  __syncthreads();

  float sum = 0.f;
  for (int i = tid; i < n; i += 256) sum += __expf(s[i] * scale - m);
#pragma unroll
  for (int off = 16; off; off >>= 1) sum += __shfl_xor(sum, off, 32);
  if ((tid & 31) == 0) red[tid >> 5] = sum;
  __syncthreads();
  sum = 0.f;
#pragma unroll
  for (int w = 0; w < 8; ++w) sum += red[w];
  const float inv = 1.0f / sum;

  for (int i = tid; i < 2048; i += 256) {
    float v = (i < n) ? __expf(s[i] * scale - m) * inv : 0.0f;
    p[i] = (__bf16)v;
  }
}

// ---------------------------------------------------------------------------
// GEMM3: out = P @ V (per batch, M=2048 rows t, N=1024 cols c, K-dim = 2048
// keys, truncated at causal boundary). B^T = Vt [c][T].
__global__ __launch_bounds__(256)
void out_gemm(const __bf16* __restrict__ P, const __bf16* __restrict__ Vt,
              float* __restrict__ out) {
  const int z = blockIdx.z;
  const __bf16* Pp = P  + (size_t)z * 2048 * 2048;
  const __bf16* V  = Vt + (size_t)z * 1024 * 2048;
  float* O = out + (size_t)z * 2048 * 1024;

  const int lane = threadIdx.x & 31;
  const int wave = threadIdx.x >> 5;
  const int wm = wave & 1, wn = wave >> 1;
  const int mBase = blockIdx.y * 128 + wm * 64;
  const int nBase = blockIdx.x * 256 + wn * 64;
  const int Klim = blockIdx.y * 128 + 128;   // keys beyond row block are zero

  const v8f zero = {0.f, 0.f, 0.f, 0.f, 0.f, 0.f, 0.f, 0.f};
  v8f acc[4][4];
#pragma unroll
  for (int i = 0; i < 4; ++i)
#pragma unroll
    for (int j = 0; j < 4; ++j) acc[i][j] = zero;

  const __bf16* A0 = Pp + (size_t)mBase * 2048;
  for (int k = 0; k < Klim; k += 32) {
    v16bf a[4];
#pragma unroll
    for (int i = 0; i < 4; ++i)
      a[i] = load_a_frag(A0 + (size_t)(i * 16) * 2048, 2048, k, lane);
#pragma unroll
    for (int j = 0; j < 4; ++j) {
      v16bf bf = load_b_frag(V + (size_t)(nBase + j * 16) * 2048, 2048, k, lane);
#pragma unroll
      for (int i = 0; i < 4; ++i) acc[i][j] = wmma_bf16(a[i], bf, acc[i][j]);
    }
  }

  const int half = lane >> 4, nl = lane & 15;
#pragma unroll
  for (int i = 0; i < 4; ++i)
#pragma unroll
    for (int j = 0; j < 4; ++j) {
      const int col = nBase + j * 16 + nl;
#pragma unroll
      for (int r = 0; r < 8; ++r) {
        const int row = mBase + i * 16 + half * 8 + r;
        O[(size_t)row * 1024 + col] = acc[i][j][r];
      }
    }
}

// ---------------------------------------------------------------------------
extern "C" void kernel_launch(void* const* d_in, const int* in_sizes, int n_in,
                              void* d_out, int out_size, void* d_ws, size_t ws_size,
                              hipStream_t stream) {
  (void)in_sizes; (void)n_in; (void)out_size; (void)ws_size;
  const float* x = (const float*)d_in[0];   // [4,2048,1024]
  const float* W = (const float*)d_in[1];   // [1024,3072]
  const float* b = (const float*)d_in[2];   // [3072]
  float* out = (float*)d_out;               // [4,2048,1024]

  char* ws = (char*)d_ws;
  // workspace layout (bytes)
  __bf16* xb = (__bf16*)(ws + 0);            //  16,777,216  x as bf16 [8192][1024]
  __bf16* Wt = (__bf16*)(ws + 16777216);     //   6,291,456  W^T bf16 [3072][1024]
  __bf16* Qb = (__bf16*)(ws + 23068672);     //  16,777,216  Q bf16 [4][2048][1024]
  __bf16* Kb = (__bf16*)(ws + 39845888);     //  16,777,216  K bf16 [4][2048][1024]
  __bf16* Vt = (__bf16*)(ws + 56623104);     //  16,777,216  V^T bf16 [4][1024][2048]
  float*  S  = (float*) (ws + 73400320);     //  67,108,864  scores f32 [4][2048][2048]
  __bf16* P  = (__bf16*)(ws + 140509184);    //  33,554,432  probs bf16 [4][2048][2048]
  // total: 174,063,616 bytes

  dim3 blk(256);
  cvt_bf16<<<8388608 / 256, blk, 0, stream>>>(x, xb, 8388608);
  cvt_Wt<<<3145728 / 256, blk, 0, stream>>>(W, Wt);
  qkv_gemm<<<dim3(12, 64), blk, 0, stream>>>(xb, Wt, b, Qb, Kb, Vt);
  scores_gemm<<<dim3(8, 16, 4), blk, 0, stream>>>(Qb, Kb, S);
  softmax_causal<<<8192, blk, 0, stream>>>(S, P);
  out_gemm<<<dim3(4, 16, 4), blk, 0, stream>>>(P, Vt, out);
}
